// ViTMultiHeadAttention_26414048870559
// MI455X (gfx1250) — matrix operand order repaired and checked by reading, not compiled, not run on hardware
//
#include <hip/hip_runtime.h>

// ---------------- problem constants ----------------
#define BATCH 64
#define SEQ   197
#define DMODEL 768
#define NHEAD 12
#define HDIM  64
#define SPAD  224          // SEQ padded to multiple of 32 (7 k-steps of 32)
#define MROWS (BATCH * SEQ)        // 12608 = 788 * 16 exactly
#define ATT_SCALE 0.125f           // 64^-0.5

// ---------------- WMMA types / helpers ----------------
typedef __bf16 v16bf __attribute__((ext_vector_type(16)));
typedef __bf16 bf16x8 __attribute__((ext_vector_type(8)));
typedef float  v8f    __attribute__((ext_vector_type(8)));

__device__ __forceinline__ unsigned short f2bf(float f) {
  unsigned u = __builtin_bit_cast(unsigned, f);
  u += 0x7FFFu + ((u >> 16) & 1u);          // round to nearest even
  return (unsigned short)(u >> 16);
}

// A-operand (16x32 bf16): per ISA, lane<16 holds K {kb..kb+7, kb+16..kb+23}
__device__ __forceinline__ v16bf load_a16(const unsigned short* p) {
  union { v16bf v; bf16x8 h[2]; } u;
  u.h[0] = *reinterpret_cast<const bf16x8*>(p);
  u.h[1] = *reinterpret_cast<const bf16x8*>(p + 16);
  return u.v;
}
// B-operand (32x16 bf16): lane<16 holds K {kb..kb+7, kb+16..kb+23} of its column
__device__ __forceinline__ v16bf load_b16(const unsigned short* p) {
  union { v16bf v; bf16x8 h[2]; } u;
  u.h[0] = *reinterpret_cast<const bf16x8*>(p);
  u.h[1] = *reinterpret_cast<const bf16x8*>(p + 16);
  return u.v;
}
__device__ __forceinline__ v8f wmma_bf16(v16bf a, v16bf b, v8f c) {
  return __builtin_amdgcn_wmma_f32_16x16x32_bf16(
      /*neg_a=*/false, a, /*neg_b=*/false, b,
      /*c_mod=*/(short)0, c, /*reuse_a=*/false, /*reuse_b=*/false);
}

// ---------------- prep kernels ----------------
__global__ void fill_zero_u32(unsigned* __restrict__ p, int n) {
  int i = blockIdx.x * blockDim.x + threadIdx.x;
  if (i < n) p[i] = 0u;
}

__global__ void cast_f32_bf16(const float* __restrict__ in,
                              unsigned short* __restrict__ out, int n) {
  int i = blockIdx.x * blockDim.x + threadIdx.x;
  if (i < n) out[i] = f2bf(in[i]);
}

// in: [cols_out][rows_out] f32 row-major ; out: [rows_out][cols_out] bf16
__global__ void transpose_f32_bf16(const float* __restrict__ in,
                                   unsigned short* __restrict__ out,
                                   int rows_out, int cols_out) {
  int i = blockIdx.x * blockDim.x + threadIdx.x;
  if (i < rows_out * cols_out) {
    int n = i / cols_out;
    int k = i - n * cols_out;
    out[i] = f2bf(in[(size_t)k * rows_out + n]);
  }
}

// ---------------- QKV projection: xb[12608x768] @ qkv_w -> Q/K/V^T ----------------
// one wave computes a 32x64 tile of the 12608x2304 result
__global__ __launch_bounds__(32) void qkv_gemm(
    const unsigned short* __restrict__ xb,   // [12608][768] bf16
    const unsigned short* __restrict__ wqt,  // [2304][768] bf16 (W^T)
    const float* __restrict__ bias,          // [2304]
    unsigned short* __restrict__ Qb,         // [B,H,SPAD,HD] bf16 (pre-scaled)
    unsigned short* __restrict__ Kb,         // [B,H,SPAD,HD] bf16
    unsigned short* __restrict__ Vt) {       // [B,H,HD,SPAD] bf16
  const int l  = threadIdx.x;
  const int lm = l & 15, lh = l >> 4;
  const int mg = blockIdx.x;   // 394 groups of 32 rows
  const int ng = blockIdx.y;   // 36 groups of 64 cols

  v8f z8 = {0.f, 0.f, 0.f, 0.f, 0.f, 0.f, 0.f, 0.f};
  v8f c[2][4];
#pragma unroll
  for (int mi = 0; mi < 2; ++mi)
#pragma unroll
    for (int ni = 0; ni < 4; ++ni) c[mi][ni] = z8;

  for (int kk = 0; kk < DMODEL / 32; ++kk) {
    const int kbA = kk * 32 + (lh << 3);
    const int kbB = kk * 32 + (lh << 3);
    v16bf a0 = load_a16(xb + (size_t)(mg * 32 + lm) * DMODEL + kbA);
    v16bf a1 = load_a16(xb + (size_t)(mg * 32 + 16 + lm) * DMODEL + kbA);
#pragma unroll
    for (int ni = 0; ni < 4; ++ni) {
      v16bf bt = load_b16(wqt + (size_t)(ng * 64 + ni * 16 + lm) * DMODEL + kbB);
      c[0][ni] = wmma_bf16(a0, bt, c[0][ni]);
      c[1][ni] = wmma_bf16(a1, bt, c[1][ni]);
    }
  }

  // epilogue: bias, scale Q, scatter into padded head-major layouts
#pragma unroll
  for (int ni = 0; ni < 4; ++ni) {
    const int col = ng * 64 + ni * 16 + lm;           // 0..2303
    const float bcol = bias[col];
    const int sec = col / DMODEL;                     // 0=Q 1=K 2=V
    const int cc  = col - sec * DMODEL;
    const int hh  = cc >> 6;
    const int dd  = cc & 63;
#pragma unroll
    for (int mi = 0; mi < 2; ++mi)
#pragma unroll
      for (int r = 0; r < 8; ++r) {
        const int row = mg * 32 + mi * 16 + (lh << 3) + r;  // < 12608
        const int bb = row / SEQ;
        const int s  = row - bb * SEQ;
        const size_t bh = (size_t)bb * NHEAD + hh;
        const float v = c[mi][ni][r] + bcol;
        if (sec == 0)      Qb[(bh * SPAD + s) * HDIM + dd] = f2bf(v * ATT_SCALE);
        else if (sec == 1) Kb[(bh * SPAD + s) * HDIM + dd] = f2bf(v);
        else               Vt[(bh * HDIM + dd) * SPAD + s] = f2bf(v);
      }
  }
}

// ---------------- fused attention: one wave per (b, h, 16-row q-tile) ----------------
__global__ __launch_bounds__(32) void attn_kernel(
    const unsigned short* __restrict__ Qb,
    const unsigned short* __restrict__ Kb,
    const unsigned short* __restrict__ Vt,
    unsigned short* __restrict__ ao) {        // [12608][768] bf16
  __shared__ float          ssc[16 * SPAD];   // score tile fp32
  __shared__ unsigned short sp[16 * SPAD];    // softmaxed P bf16

  const int l  = threadIdx.x;
  const int lm = l & 15, lh = l >> 4;
  const int qt = blockIdx.x;   // 13 q-tiles
  const int h  = blockIdx.y;   // 12 heads
  const int b  = blockIdx.z;   // 64 batch

  const size_t bh = (size_t)b * NHEAD + h;
  const unsigned short* Qp = Qb + bh * SPAD * HDIM;
  const unsigned short* Kp = Kb + bh * SPAD * HDIM;
  const unsigned short* Vp = Vt + bh * HDIM * SPAD;

  // Q tile (16 rows x 64) as two A-operands
  v16bf qa[2];
#pragma unroll
  for (int kk = 0; kk < 2; ++kk)
    qa[kk] = load_a16(Qp + (size_t)(qt * 16 + lm) * HDIM + kk * 32 + (lh << 3));

  v8f z8 = {0.f, 0.f, 0.f, 0.f, 0.f, 0.f, 0.f, 0.f};

  // scores = (Q*scale) @ K^T : 13 n-tiles cover cols 0..207
  for (int nt = 0; nt < 13; ++nt) {
    v8f cs = z8;
#pragma unroll
    for (int kk = 0; kk < 2; ++kk) {
      v16bf bt = load_b16(Kp + (size_t)(nt * 16 + lm) * HDIM + kk * 32 + (lh << 3));
      cs = wmma_bf16(qa[kk], bt, cs);
    }
#pragma unroll
    for (int r = 0; r < 8; ++r)
      ssc[((lh << 3) + r) * SPAD + nt * 16 + lm] = cs[r];
  }
  __syncthreads();

  // softmax over 197 valid cols; 2 lanes per row, halves of 112 cols each
  {
    const int row = lm;
    const int c0  = lh * 112;
    float mx = -1e30f;
    for (int j = 0; j < 112; ++j) {
      int col = c0 + j;
      if (col < SEQ) mx = fmaxf(mx, ssc[row * SPAD + col]);
    }
    mx = fmaxf(mx, __shfl_xor(mx, 16, 32));
    float sum = 0.f;
    for (int j = 0; j < 112; ++j) {
      int col = c0 + j;
      if (col < SEQ) sum += __expf(ssc[row * SPAD + col] - mx);
    }
    sum += __shfl_xor(sum, 16, 32);
    const float inv = 1.0f / sum;
    for (int j = 0; j < 112; ++j) {
      int col = c0 + j;
      sp[row * SPAD + col] =
          (col < SEQ) ? f2bf(__expf(ssc[row * SPAD + col] - mx) * inv)
                      : (unsigned short)0;
    }
  }
  __syncthreads();

  // out_tile = P @ V   (A from LDS, B from V^T; 7 k-steps of 32 over SPAD)
  v8f co[4];
#pragma unroll
  for (int nt = 0; nt < 4; ++nt) co[nt] = z8;
  for (int kk = 0; kk < SPAD / 32; ++kk) {
    v16bf pa = load_a16(sp + lm * SPAD + kk * 32 + (lh << 3));
#pragma unroll
    for (int nt = 0; nt < 4; ++nt) {
      v16bf bt = load_b16(Vp + (size_t)(nt * 16 + lm) * SPAD + kk * 32 + (lh << 3));
      co[nt] = wmma_bf16(pa, bt, co[nt]);
    }
  }

#pragma unroll
  for (int nt = 0; nt < 4; ++nt)
#pragma unroll
    for (int r = 0; r < 8; ++r) {
      const int s = qt * 16 + (lh << 3) + r;
      if (s < SEQ)
        ao[((size_t)b * SEQ + s) * DMODEL + h * HDIM + nt * 16 + lm] =
            f2bf(co[nt][r]);
    }
}

// ---------------- output projection: ao @ proj_w + proj_b -> fp32 ----------------
__global__ __launch_bounds__(32) void proj_gemm(
    const unsigned short* __restrict__ ab,   // [12608][768] bf16
    const unsigned short* __restrict__ wpt,  // [768][768] bf16 (W^T)
    const float* __restrict__ bias,          // [768]
    float* __restrict__ out) {               // [12608][768] f32
  const int l  = threadIdx.x;
  const int lm = l & 15, lh = l >> 4;
  const int mg = blockIdx.x;   // 394
  const int ng = blockIdx.y;   // 12

  v8f z8 = {0.f, 0.f, 0.f, 0.f, 0.f, 0.f, 0.f, 0.f};
  v8f c[2][4];
#pragma unroll
  for (int mi = 0; mi < 2; ++mi)
#pragma unroll
    for (int ni = 0; ni < 4; ++ni) c[mi][ni] = z8;

  for (int kk = 0; kk < DMODEL / 32; ++kk) {
    const int kbA = kk * 32 + (lh << 3);
    const int kbB = kk * 32 + (lh << 3);
    v16bf a0 = load_a16(ab + (size_t)(mg * 32 + lm) * DMODEL + kbA);
    v16bf a1 = load_a16(ab + (size_t)(mg * 32 + 16 + lm) * DMODEL + kbA);
#pragma unroll
    for (int ni = 0; ni < 4; ++ni) {
      v16bf bt = load_b16(wpt + (size_t)(ng * 64 + ni * 16 + lm) * DMODEL + kbB);
      c[0][ni] = wmma_bf16(a0, bt, c[0][ni]);
      c[1][ni] = wmma_bf16(a1, bt, c[1][ni]);
    }
  }

#pragma unroll
  for (int ni = 0; ni < 4; ++ni) {
    const int col = ng * 64 + ni * 16 + lm;
    const float bcol = bias[col];
#pragma unroll
    for (int mi = 0; mi < 2; ++mi)
#pragma unroll
      for (int r = 0; r < 8; ++r) {
        const int row = mg * 32 + mi * 16 + (lh << 3) + r;
        out[(size_t)row * DMODEL + col] = c[mi][ni][r] + bcol;
      }
  }
}

// ---------------- host-side orchestration ----------------
extern "C" void kernel_launch(void* const* d_in, const int* in_sizes, int n_in,
                              void* d_out, int out_size, void* d_ws, size_t ws_size,
                              hipStream_t stream) {
  const float* x      = (const float*)d_in[0];   // [64,197,768]
  const float* qkv_w  = (const float*)d_in[1];   // [768,2304]
  const float* qkv_b  = (const float*)d_in[2];   // [2304]
  const float* proj_w = (const float*)d_in[3];   // [768,768]
  const float* proj_b = (const float*)d_in[4];   // [768]
  float* out = (float*)d_out;

  // workspace layout (all bf16 as unsigned short); each region 256B-aligned
  const size_t N_XB    = (size_t)MROWS * DMODEL;            // 9,682,944
  const size_t N_WQT   = (size_t)3 * DMODEL * DMODEL;       // 1,769,472
  const size_t N_WPT   = (size_t)DMODEL * DMODEL;           //   589,824
  const size_t N_HEADT = (size_t)BATCH * NHEAD * SPAD * HDIM; // 11,010,048
  const size_t N_AO    = (size_t)MROWS * DMODEL;

  char* w = (char*)d_ws;
  auto take = [&](size_t bytes) {
    char* r = w;
    w += (bytes + 255) & ~(size_t)255;
    return r;
  };
  unsigned short* xb  = (unsigned short*)take(N_XB * 2);
  unsigned short* wqt = (unsigned short*)take(N_WQT * 2);
  unsigned short* wpt = (unsigned short*)take(N_WPT * 2);
  unsigned short* Qb  = (unsigned short*)take(N_HEADT * 2);  // these three are
  unsigned short* Kb  = (unsigned short*)take(N_HEADT * 2);  // contiguous (sizes
  unsigned short* Vt  = (unsigned short*)take(N_HEADT * 2);  // are 256B multiples)
  unsigned short* ao  = (unsigned short*)take(N_AO * 2);

  // 1) zero padded Q/K/V^T region (S rows 197..223 must be clean zeros)
  {
    int ndw = (int)(3 * N_HEADT / 2);
    fill_zero_u32<<<(ndw + 255) / 256, 256, 0, stream>>>((unsigned*)Qb, ndw);
  }
  // 2) cast x to bf16
  cast_f32_bf16<<<((int)N_XB + 255) / 256, 256, 0, stream>>>(x, xb, (int)N_XB);
  // 3) transpose+cast weights (so WMMA B-operand loads are contiguous)
  transpose_f32_bf16<<<((int)N_WQT + 255) / 256, 256, 0, stream>>>(
      qkv_w, wqt, 3 * DMODEL, DMODEL);
  transpose_f32_bf16<<<((int)N_WPT + 255) / 256, 256, 0, stream>>>(
      proj_w, wpt, DMODEL, DMODEL);
  // 4) QKV projection + scatter to head-major padded layouts
  qkv_gemm<<<dim3(MROWS / 32, (3 * DMODEL) / 64), 32, 0, stream>>>(
      xb, wqt, qkv_b, Qb, Kb, Vt);
  // 5) fused attention (softmax never leaves LDS)
  attn_kernel<<<dim3(13, NHEAD, BATCH), 32, 0, stream>>>(Qb, Kb, Vt, ao);
  // 6) output projection + bias -> fp32
  proj_gemm<<<dim3(MROWS / 32, DMODEL / 64), 32, 0, stream>>>(
      ao, wpt, proj_b, out);
}
